// LICWorldModel_30709016166896
// MI455X (gfx1250) — matrix operand order repaired and checked by reading, not compiled, not run on hardware
//
#include <hip/hip_runtime.h>
#include <hip/hip_bf16.h>
#include <math.h>

// ---------------- dims ----------------
// B=8, T=17, N=256, D=8, A=4, E=8, H=128, C=64, CH=4, P=16, HOR=16
// GRU fused GEMM: M = B*N = 2048, Kpad = 288 (x8|a4|msg128|h128|pad20), Ncols = 512
//   cols [0,128)   : r-gate  (Wih_r + Whh_r fused, bias bih_r+bhh_r)
//   cols [128,256) : z-gate  (fused)
//   cols [256,384) : inn     (Wih_n only, bias bih_n)
//   cols [384,512) : hn      (Whh_n only, bias bhh_n)

constexpr int KP = 288;      // padded K
constexpr int NT = 32;       // n-tiles (512/16)
constexpr int KT = 9;        // k-tiles (288/32)
constexpr int WPACK_ELEMS = KT * NT * 32 * 16;   // 147456 bf16

// output offsets (floats)
constexpr int FAST_OFF = 8 * 16 * 256 * 8;                  // 262144
constexpr int CT_OFF   = FAST_OFF + 8 * 16 * 256 * 128;     // 4456448
constexpr int ET_OFF   = CT_OFF + 8 * 16 * 64;              // 4464640

// workspace offsets (floats; wpack is bf16 but sized in float slots)
constexpr int WS_BIAS = 73728;            // 512 f
constexpr int WS_C    = 74240;            // 8*64 f
constexpr int WS_LAW  = 74752;            // 8*12 f  (rho[4],beta[4],tau[4] per b)
constexpr int WS_H    = 74848;            // 8*256*128 f (16B aligned: 74848*4 % 16 == 0)
constexpr int WS_MSG  = WS_H + 262144;    // 8*256*128 f

typedef __bf16 bf16x8  __attribute__((ext_vector_type(8)));
typedef __bf16 bf16x16 __attribute__((ext_vector_type(16)));
typedef float  f32x8   __attribute__((ext_vector_type(8)));

struct P {
  const float *obs, *act, *evt;
  const float *bank_Wu, *bank_bu, *bank_logsig, *bank_mu;
  const float *enc_W1, *enc_W2, *enc_b1, *enc_b2;
  const float *gru_Whh, *gru_Wih, *gru_bhh, *gru_bih;
  const float *law_Wb, *law_Wr, *law_Wt, *law_bb, *law_br, *law_bt;
  const float *msg_W, *msg_b, *pool_W, *pool_b, *pred_W, *pred_b;
  const float *tr_Wbar, *tr_Wd, *tr_We, *tr_Wo, *tr_bbar, *tr_bd, *tr_be, *tr_bo;
  __bf16 *wpack;
  float *bias512, *c_cur, *law, *h, *msgbuf, *out;
};

__device__ __forceinline__ float sigm_(float x) { return 1.0f / (1.0f + __expf(-x)); }
__device__ __forceinline__ float softplus_(float x) { return (x > 20.0f) ? x : log1pf(__expf(x)); }

// rho/beta/tau from climate state c (shared mem) -> p.law[b*12 + ...]
__device__ __forceinline__ void law_from_c(const P& p, const float* cS, int b, int tid) {
  if (tid < 12) {
    int ch = tid & 3, which = tid >> 2;            // 0:rho 1:beta 2:tau
    const float* W = (which == 0) ? p.law_Wr : (which == 1) ? p.law_Wb : p.law_Wt;
    float a = (which == 0) ? p.law_br[ch] : (which == 1) ? p.law_bb[ch] : p.law_bt[ch];
    for (int j = 0; j < 64; ++j) a += cS[j] * W[j * 4 + ch];
    float v = (which == 1) ? a : softplus_(a);
    if (which == 2) v += 0.1f;
    p.law[b * 12 + (which << 2) + ch] = v;
  }
}

// ---------------- one-time: pack GRU weights into bf16 WMMA B-fragment order ----------------
// B 32x16 bf16 fragment, contiguous k order this code uses for both A and B: element(k_local,n_local): lane = n_local + 16*(k_local/16), elem = k_local%16
__global__ __launch_bounds__(256) void k_pack(P p) {
  int idx = blockIdx.x * 256 + threadIdx.x;
  if (idx < WPACK_ELEMS) {
    int e    = idx & 15;
    int lane = (idx >> 4) & 31;
    int nt   = (idx >> 9) & 31;
    int kt   = idx >> 14;
    int k    = kt * 32 + ((lane >> 4) << 4) + e;       // B-operand layout
    int col  = (nt << 4) + (lane & 15);
    int gb   = col >> 7, cc = col & 127;
    float v = 0.0f;
    if (gb <= 1) {                                     // r / z fused
      int oc = (gb << 7) + cc;
      if (k < 140)      v = p.gru_Wih[k * 384 + oc];
      else if (k < 268) v = p.gru_Whh[(k - 140) * 384 + oc];
    } else if (gb == 2) {                              // inn (input path only)
      if (k < 140)      v = p.gru_Wih[k * 384 + 256 + cc];
    } else {                                           // hn (hidden path only)
      if (k >= 140 && k < 268) v = p.gru_Whh[(k - 140) * 384 + 256 + cc];
    }
    p.wpack[idx] = (__bf16)v;
  }
  if (idx < 512) {
    int cc = idx & 127, gb = idx >> 7;
    float v;
    if (gb == 0)      v = p.gru_bih[cc] + p.gru_bhh[cc];
    else if (gb == 1) v = p.gru_bih[128 + cc] + p.gru_bhh[128 + cc];
    else if (gb == 2) v = p.gru_bih[256 + cc];
    else              v = p.gru_bhh[256 + cc];
    p.bias512[idx] = v;
  }
}

// ---------------- climate encoder + h0 zero + law(c0) ----------------
__global__ __launch_bounds__(256) void k_enc(P p) {
  int b = blockIdx.x, tid = threadIdx.x;
  __shared__ float feat[20], hidS[128], c0S[64];
  for (int idx = tid; idx < 256 * 128; idx += 256) p.h[(b << 15) + idx] = 0.0f;
  if (tid < 8) {
    float a = 0.0f;
    for (int t = 0; t < 17; ++t)
      for (int i = 0; i < 256; ++i) a += p.obs[((((b * 17 + t) * 256) + i) << 3) + tid];
    feat[tid] = a * (1.0f / (17.0f * 256.0f));
  } else if (tid < 12) {
    int d = tid - 8; float a = 0.0f;
    for (int t = 0; t < 17; ++t)
      for (int i = 0; i < 256; ++i) a += p.act[((((b * 17 + t) * 256) + i) << 2) + d];
    feat[tid] = a * (1.0f / (17.0f * 256.0f));
  } else if (tid < 20) {
    int e = tid - 12; float a = 0.0f;
    for (int t = 0; t < 17; ++t) a += p.evt[((b * 17 + t) << 3) + e];
    feat[tid] = a * (1.0f / 17.0f);
  }
  __syncthreads();
  if (tid < 128) {
    float a = p.enc_b1[tid];
    for (int j = 0; j < 20; ++j) a += feat[j] * p.enc_W1[j * 128 + tid];
    hidS[tid] = fmaxf(a, 0.0f);
  }
  __syncthreads();
  if (tid < 64) {
    float a = p.enc_b2[tid];
    for (int j = 0; j < 128; ++j) a += hidS[j] * p.enc_W2[j * 64 + tid];
    float c0 = tanhf(a);
    c0S[tid] = c0;
    p.c_cur[(b << 6) + tid] = c0;
  }
  __syncthreads();
  law_from_c(p, c0S, b, tid);
}

// ---------------- per-step: pairwise interactions + message MLP ----------------
__global__ __launch_bounds__(256) void k_pair(P p, int s) {
  int b = blockIdx.x >> 8, i = blockIdx.x & 255;
  int tid = threadIdx.x, lane = tid & 31, wave = tid >> 5;
  __shared__ float xi[8], muS[16], siS[16], WuS[64], buS[4], lawS[12];
  __shared__ float red[8][32], aggS[32];
  if (tid < 8) xi[tid] = p.obs[((((b * 17 + s) * 256) + i) << 3) + tid];
  if (tid >= 32 && tid < 48) { int t = tid - 32; muS[t] = p.bank_mu[t]; siS[t] = __expf(-p.bank_logsig[t]); }
  if (tid >= 64 && tid < 128) WuS[tid - 64] = p.bank_Wu[tid - 64];
  if (tid >= 128 && tid < 132) buS[tid - 128] = p.bank_bu[tid - 128];
  if (tid >= 160 && tid < 172) lawS[tid - 160] = p.law[b * 12 + (tid - 160)];
  __syncthreads();
  int j = tid;
  const float* xj = &p.obs[((((b * 17 + s) * 256) + j) << 3)];
  float ph[8];
  #pragma unroll
  for (int d = 0; d < 8; ++d) ph[d] = xi[d] - xj[d];
  float dij = sqrtf(ph[0] * ph[0] + ph[1] * ph[1]);
  float rbf[16];
  #pragma unroll
  for (int q = 0; q < 16; ++q) { float t = (dij - muS[q]) * siS[q]; rbf[q] = __expf(-0.5f * t * t); }
  float wc[4];
  #pragma unroll
  for (int c = 0; c < 4; ++c) {
    float a = buS[c];
    #pragma unroll
    for (int q = 0; q < 16; ++q) a += rbf[q] * WuS[q * 4 + c];
    float u   = sigm_(a);
    float chi = sigm_(lawS[4 + c] - dij);
    float gam = lawS[c] * u * __expf(-dij / lawS[8 + c]);
    wc[c] = gam * chi;
  }
  #pragma unroll
  for (int c = 0; c < 4; ++c) {
    #pragma unroll
    for (int d = 0; d < 8; ++d) {
      float v = wc[c] * ph[d];
      v += __shfl_xor(v, 16, 32); v += __shfl_xor(v, 8, 32);
      v += __shfl_xor(v, 4, 32);  v += __shfl_xor(v, 2, 32); v += __shfl_xor(v, 1, 32);
      if (lane == 0) red[wave][c * 8 + d] = v;
    }
  }
  __syncthreads();
  if (tid < 32) {
    float a = 0.0f;
    #pragma unroll
    for (int w2 = 0; w2 < 8; ++w2) a += red[w2][tid];
    aggS[tid] = a;
  }
  __syncthreads();
  if (tid < 128) {
    float a = p.msg_b[tid];
    #pragma unroll
    for (int t = 0; t < 32; ++t) a += aggS[t] * p.msg_W[t * 128 + tid];
    p.msgbuf[((b << 8) + i) * 128 + tid] = tanhf(a);
  }
}

// ---------------- per-step: fused GRU GEMM (bf16 WMMA) + gates + x_pred ----------------
__global__ __launch_bounds__(256) void k_gru(P p, int s) {
  int tid = threadIdx.x, lane = tid & 31, wave = tid >> 5;
  int row0 = blockIdx.x << 4;                       // 16-row M tile
  __shared__ __bf16 Ap[16 * KP];                    // 9 KB bf16 A-panel
  __shared__ float hnewS[16 * 128];                 // 8 KB h_new staging
  for (int idx = tid; idx < 16 * KP; idx += 256) {
    int m = idx / KP, k = idx - m * KP;
    int row = row0 + m, b = row >> 8, i = row & 255;
    float v;
    if (k < 8)        v = p.obs[((((b * 17 + s) * 256) + i) << 3) + k];
    else if (k < 12)  v = p.act[((((b * 17 + s) * 256) + i) << 2) + (k - 8)];
    else if (k < 140) v = p.msgbuf[(row << 7) + (k - 12)];
    else if (k < 268) v = p.h[(row << 7) + (k - 140)];
    else              v = 0.0f;
    Ap[idx] = (__bf16)v;
  }
  __syncthreads();
  f32x8 acc0 = {0,0,0,0,0,0,0,0}, acc1 = {0,0,0,0,0,0,0,0};
  f32x8 acc2 = {0,0,0,0,0,0,0,0}, acc3 = {0,0,0,0,0,0,0,0};
  int ma = lane & 15, hf = lane >> 4;
  #pragma unroll
  for (int kt = 0; kt < KT; ++kt) {
    // A 16x32 bf16 fragment: elem e -> k = kt*32 + hf*16 + e (two contiguous 16B chunks)
    bf16x8 lo = *reinterpret_cast<const bf16x8*>(&Ap[ma * KP + kt * 32 + hf * 16]);
    bf16x8 hi = *reinterpret_cast<const bf16x8*>(&Ap[ma * KP + kt * 32 + hf * 16 + 8]);
    bf16x16 a = __builtin_shufflevector(lo, hi, 0,1,2,3,4,5,6,7,8,9,10,11,12,13,14,15);
    const __bf16* wb = &p.wpack[((kt * NT) * 32) * 16];
    bf16x16 b0 = *reinterpret_cast<const bf16x16*>(&wb[((wave      ) * 32 + lane) * 16]);
    bf16x16 b1 = *reinterpret_cast<const bf16x16*>(&wb[((wave +  8) * 32 + lane) * 16]);
    bf16x16 b2 = *reinterpret_cast<const bf16x16*>(&wb[((wave + 16) * 32 + lane) * 16]);
    bf16x16 b3 = *reinterpret_cast<const bf16x16*>(&wb[((wave + 24) * 32 + lane) * 16]);
    acc0 = __builtin_amdgcn_wmma_f32_16x16x32_bf16(false, a, false, b0, (short)0, acc0, false, false);
    acc1 = __builtin_amdgcn_wmma_f32_16x16x32_bf16(false, a, false, b1, (short)0, acc1, false, false);
    acc2 = __builtin_amdgcn_wmma_f32_16x16x32_bf16(false, a, false, b2, (short)0, acc2, false, false);
    acc3 = __builtin_amdgcn_wmma_f32_16x16x32_bf16(false, a, false, b3, (short)0, acc3, false, false);
  }
  // gates: wave w owns gate channels c = 16w + lane%16 for all 16 rows
  int c = (wave << 4) + (lane & 15);
  float br_ = p.bias512[c], bz_ = p.bias512[128 + c];
  float bi_ = p.bias512[256 + c], bh_ = p.bias512[384 + c];
  #pragma unroll
  for (int e = 0; e < 8; ++e) {
    int m = e + (hf << 3);                          // D layout: m = vgpr + 8*(lane/16)
    int row = row0 + m, b = row >> 8, i = row & 255;
    float r   = sigm_(acc0[e] + br_);
    float z   = sigm_(acc1[e] + bz_);
    float nst = tanhf(acc2[e] + bi_ + r * (acc3[e] + bh_));
    float hp  = p.h[(row << 7) + c];
    float hn  = (1.0f - z) * nst + z * hp;
    p.h[(row << 7) + c] = hn;
    p.out[FAST_OFF + ((((b << 4) + s) << 8) + i) * 128 + c] = hn;
    hnewS[(m << 7) + c] = hn;
  }
  __syncthreads();
  if (tid < 128) {
    int m = tid >> 3, d = tid & 7;
    float a = p.pred_b[d];
    for (int cc = 0; cc < 128; ++cc) a += hnewS[(m << 7) + cc] * p.pred_W[(cc << 3) + d];
    int row = row0 + m, b = row >> 8, i = row & 255;
    float xp = p.obs[((((b * 17 + s) * 256) + i) << 3) + d] + a;
    p.out[((((b << 4) + s) << 8) + i) * 8 + d] = xp;
  }
}

// ---------------- per-step: pool + climate transition + law(c_next) ----------------
__global__ __launch_bounds__(256) void k_climate(P p, int s) {
  int b = blockIdx.x, tid = threadIdx.x;
  __shared__ float meanS[136], qS[128], ti[200], dcS[64], cbS[64], oeS[2], cnS[64];
  if (tid < 128) {
    float a = 0.0f;
    for (int i = 0; i < 256; ++i) a += p.out[FAST_OFF + ((((b << 4) + s) << 8) + i) * 128 + tid];
    meanS[tid] = a * (1.0f / 256.0f);
  } else if (tid < 136) {
    int d = tid - 128; float a = 0.0f;
    for (int i = 0; i < 256; ++i) a += p.out[((((b << 4) + s) << 8) + i) * 8 + d];
    meanS[tid] = a * (1.0f / 256.0f);
  }
  __syncthreads();
  if (tid < 128) {
    float a = p.pool_b[tid];
    for (int j = 0; j < 136; ++j) a += meanS[j] * p.pool_W[j * 128 + tid];
    qS[tid] = tanhf(a);
  }
  __syncthreads();
  if (tid < 64)        ti[tid] = p.c_cur[(b << 6) + tid];
  else if (tid < 192)  ti[tid] = qS[tid - 64];
  else if (tid < 200)  ti[tid] = p.evt[((b * 17 + s) << 3) + (tid - 192)];
  __syncthreads();
  if (tid < 64) {
    float a = p.tr_bd[tid];
    for (int j = 0; j < 200; ++j) a += ti[j] * p.tr_Wd[j * 64 + tid];
    dcS[tid] = tanhf(a);
  } else if (tid < 128) {
    int o = tid - 64; float a = p.tr_bbar[o];
    for (int j = 0; j < 200; ++j) a += ti[j] * p.tr_Wbar[j * 64 + o];
    cbS[o] = tanhf(a);
  } else if (tid == 128) {
    float a = p.tr_bo[0];
    for (int j = 0; j < 200; ++j) a += ti[j] * p.tr_Wo[j];
    oeS[0] = 4.0f * sigm_(a);                        // omega
  } else if (tid == 129) {
    float a = p.tr_be[0];
    for (int j = 0; j < 200; ++j) a += ti[j] * p.tr_We[j];
    float eta = sigm_(a);
    oeS[1] = eta;
    p.out[ET_OFF + (b << 4) + s] = eta;
  }
  __syncthreads();
  if (tid < 64) {
    float cs = ti[tid] + 0.1f * dcS[tid];
    float cn = cs + oeS[1] * oeS[0] * (cbS[tid] - cs);
    p.c_cur[(b << 6) + tid] = cn;
    p.out[CT_OFF + (((b << 4) + s) << 6) + tid] = cn;
    cnS[tid] = cn;
  }
  __syncthreads();
  law_from_c(p, cnS, b, tid);                        // law params for step s+1
}

extern "C" void kernel_launch(void* const* d_in, const int* in_sizes, int n_in,
                              void* d_out, int out_size, void* d_ws, size_t ws_size,
                              hipStream_t stream) {
  (void)in_sizes; (void)n_in; (void)out_size; (void)ws_size;
  P p;
  // top-level dict order, then params leaves in ASCII-sorted (jax pytree) order
  p.obs         = (const float*)d_in[0];
  p.act         = (const float*)d_in[1];
  p.evt         = (const float*)d_in[2];
  p.bank_Wu     = (const float*)d_in[3];
  p.bank_bu     = (const float*)d_in[4];
  p.bank_logsig = (const float*)d_in[5];
  p.bank_mu     = (const float*)d_in[6];
  p.enc_W1      = (const float*)d_in[7];
  p.enc_W2      = (const float*)d_in[8];
  p.enc_b1      = (const float*)d_in[9];
  p.enc_b2      = (const float*)d_in[10];
  p.gru_Whh     = (const float*)d_in[11];
  p.gru_Wih     = (const float*)d_in[12];
  p.gru_bhh     = (const float*)d_in[13];
  p.gru_bih     = (const float*)d_in[14];
  p.law_Wb      = (const float*)d_in[15];
  p.law_Wr      = (const float*)d_in[16];
  p.law_Wt      = (const float*)d_in[17];
  p.law_bb      = (const float*)d_in[18];
  p.law_br      = (const float*)d_in[19];
  p.law_bt      = (const float*)d_in[20];
  p.msg_W       = (const float*)d_in[21];
  p.msg_b       = (const float*)d_in[22];
  p.pool_W      = (const float*)d_in[23];
  p.pool_b      = (const float*)d_in[24];
  p.pred_W      = (const float*)d_in[25];
  p.pred_b      = (const float*)d_in[26];
  p.tr_Wbar     = (const float*)d_in[27];
  p.tr_Wd       = (const float*)d_in[28];
  p.tr_We       = (const float*)d_in[29];
  p.tr_Wo       = (const float*)d_in[30];
  p.tr_bbar     = (const float*)d_in[31];
  p.tr_bd       = (const float*)d_in[32];
  p.tr_be       = (const float*)d_in[33];
  p.tr_bo       = (const float*)d_in[34];

  float* wsf = (float*)d_ws;
  p.wpack   = (__bf16*)d_ws;
  p.bias512 = wsf + WS_BIAS;
  p.c_cur   = wsf + WS_C;
  p.law     = wsf + WS_LAW;
  p.h       = wsf + WS_H;
  p.msgbuf  = wsf + WS_MSG;
  p.out     = (float*)d_out;

  k_pack<<<(WPACK_ELEMS + 255) / 256, 256, 0, stream>>>(p);
  k_enc<<<8, 256, 0, stream>>>(p);
  for (int s = 0; s < 16; ++s) {
    k_pair<<<2048, 256, 0, stream>>>(p, s);
    k_gru<<<128, 256, 0, stream>>>(p, s);
    k_climate<<<8, 256, 0, stream>>>(p, s);
  }
}
